// ModelSpectral_39702677684379
// MI455X (gfx1250) — matrix operand order repaired and checked by reading, not compiled, not run on hardware
//
#include <hip/hip_runtime.h>
#include <hip/hip_fp16.h>
#include <math.h>

// ---------------------------------------------------------------------------
// MI455X (gfx1250) hierarchical graph-SAGE network.
//
// Roofline: level-0 feature map (524288x32 f32 = 64MB) and the f16 aggregate
// (32MB) are L2-resident (192MB L2). The op is bound by edge scatter traffic,
// so the segment-sum uses global_atomic_pk_add_f16 (half the atomic ops of
// f32, matches the f16 WMMA input precision). All [*,32]x[32,*] GEMMs use
// v_wmma_f32_16x16x32_f16. Weights are staged once per workgroup into LDS by
// the Tensor Data Mover (tensor_load_to_lds + s_wait_tensorcnt). tanh uses
// the CDNA5 hardware v_tanh_f32 TRANS op (with v_nop per TRANS hazard rule).
// ---------------------------------------------------------------------------

typedef __attribute__((ext_vector_type(16))) _Float16 v16h;
typedef __attribute__((ext_vector_type(8)))  float    v8f;
typedef unsigned int u32x4 __attribute__((ext_vector_type(4)));
typedef int          i32x4 __attribute__((ext_vector_type(4)));
typedef int          i32x8 __attribute__((ext_vector_type(8)));

#define HID 32

// ------------------------- CDNA5 helpers -----------------------------------

__device__ __forceinline__ float htanh(float x) {
  float r;
  // hardware tanh (TRANS op); v_nop satisfies the 1-op TRANS result hazard
  asm volatile("v_tanh_f32 %0, %1\n\tv_nop" : "=v"(r) : "v"(x));
  return r;
}

// TDM: DMA a rows x cols f32 matrix (row-major, contiguous) global -> LDS.
// D# per CDNA5 ISA: group0 {count=1, lds_addr,
// global_addr[56:0], type=2}; group1 {data_size=4B, tensor_dim0=cols,
// tensor_dim1=rows, tile_dim0=cols, tile_dim1=rows, dim0_stride=cols}.
__device__ __forceinline__ void tdm_load_2d_f32(const float* gsrc, const void* ldst,
                                                unsigned rows, unsigned cols) {
  unsigned long long ga = (unsigned long long)(uintptr_t)gsrc;
  unsigned lo = (unsigned)(unsigned long long)(uintptr_t)ldst;  // LDS offset = addr[31:0]
  u32x4 g0;
  g0[0] = 1u;                                                  // count=1, user mode
  g0[1] = lo;                                                  // lds_addr
  g0[2] = (unsigned)ga;                                        // global_addr[31:0]
  g0[3] = (unsigned)((ga >> 32) & 0x01ffffffull) | 0x80000000u; // ga[56:32] | type=2
  i32x8 g1;
  g1[0] = 0x00020000;            // data_size = 2 (4 bytes); no pad/iterate/multicast
  g1[1] = (int)(cols << 16);     // tensor_dim0[15:0] in bits[79:64... i.e. d1[31:16]]
  g1[2] = (int)(rows << 16);     // tensor_dim1[15:0] in d2[31:16]
  g1[3] = (int)(cols << 16);     // tile_dim0 in d3[31:16]
  g1[4] = (int)rows;             // tile_dim1 (tile_dim2 = 0)
  g1[5] = (int)cols;             // tensor_dim0_stride[31:0]
  g1[6] = 0;
  g1[7] = 0;
  i32x4 z4 = {0, 0, 0, 0};
#if __clang_major__ >= 23
  i32x8 z8 = {0, 0, 0, 0, 0, 0, 0, 0};
  __builtin_amdgcn_tensor_load_to_lds(g0, g1, z4, z4, z8, 0);
#else
  __builtin_amdgcn_tensor_load_to_lds(g0, g1, z4, z4, 0);
#endif
}

// ---- WMMA fragment helpers (layouts per CDNA5 ISA §7.12.2) -----

// A fragment, 16x32 f16 from row-major f32 [16 x ld] tile.
__device__ __forceinline__ v16h load_a32(const float* __restrict__ x,
                                         int row0, int ld, int lane) {
  int r = lane & 15;
  int koff = (lane >> 4) << 3;
  const float* p = x + (size_t)(row0 + r) * ld;
  v16h a;
#pragma unroll
  for (int i = 0; i < 8; ++i) {
    a[i]     = (_Float16)p[koff + i];
    a[i + 8] = (_Float16)p[16 + koff + i];
  }
  return a;
}

// B fragment, KxN from row-major f32 W[k*ld + n] (K >= kmax zero-padded).
__device__ __forceinline__ v16h load_b(const float* W, int ld,
                                       int col0, int lane, int kmax) {
  int n = col0 + (lane & 15);
  int kb = (lane >> 4) << 3;
  v16h b;
#pragma unroll
  for (int e = 0; e < 16; ++e) {
    int k = (e < 8) ? (kb + e) : (16 + kb + (e - 8));
    b[e] = (k < kmax) ? (_Float16)W[k * ld + n] : (_Float16)0.0f;
  }
  return b;
}

// C/D 16x16 f32: N = lane&15 (+col0), M = 8*(lane>>4) + r.
__device__ __forceinline__ void store_c_tanh(float* __restrict__ out, int ld,
                                             int row0, int col0, int lane, v8f c) {
  int n = col0 + (lane & 15);
  int mhi = (lane >> 4) << 3;
#pragma unroll
  for (int r = 0; r < 8; ++r)
    out[(size_t)(row0 + mhi + r) * ld + n] = htanh(c[r]);
}

// ---------------------------- kernels --------------------------------------

__global__ void zero_kernel(float* __restrict__ p, long long n) {
  long long i = (long long)blockIdx.x * blockDim.x + threadIdx.x;
  if (i < n) p[i] = 0.0f;
}

// Coarsest level: x = eye(2); coarse_edges = [[0,1],[1,0]] (constant).
__global__ void coarse_kernel(const float* __restrict__ Wc_l,
                              const float* __restrict__ Wc_r,
                              const float* __restrict__ bc,
                              float* __restrict__ out) {
  int j = threadIdx.x;
  if (j < HID) {
    out[j]       = htanh(Wc_l[HID + j] + Wc_r[j]       + bc[j]);
    out[HID + j] = htanh(Wc_l[j]       + Wc_r[HID + j] + bc[j]);
  }
}

// Unpool: dst[i,:] = src[inv[i],:]. 8 threads/node, float4 per thread.
__global__ __launch_bounds__(256)
void gather_kernel(const float* __restrict__ src, const int* __restrict__ inv,
                   float* __restrict__ dst, int n) {
  int t = blockIdx.x * blockDim.x + threadIdx.x;
  int i = t >> 3;
  if (i >= n) return;
  int c = (t & 7) << 2;
  int j = inv[i];
  float4 v = *(const float4*)(src + (size_t)j * HID + c);
  *(float4*)(dst + (size_t)i * HID + c) = v;
}

// Segment-sum via packed f16 atomics (global_atomic_pk_add_f16, L2-resident).
// 4 threads per edge, 8 features each -> 4 pk atomics per thread.
__global__ __launch_bounds__(256)
void aggregate_kernel(const float* __restrict__ x, const int* __restrict__ edges,
                      int E, __half2* __restrict__ agg, float* __restrict__ deg) {
  int t = blockIdx.x * blockDim.x + threadIdx.x;
  int e = t >> 2;
  if (e >= E) return;
  int c = (t & 3) << 3;
  int src = edges[e];
  int dst = edges[E + e];
  const float4* xs = (const float4*)(x + (size_t)src * HID + c);
  float4 v0 = xs[0];
  float4 v1 = xs[1];
  __half2* ad = agg + (((size_t)dst * HID + c) >> 1);
  atomicAdd(ad + 0, __floats2half2_rn(v0.x, v0.y));
  atomicAdd(ad + 1, __floats2half2_rn(v0.z, v0.w));
  atomicAdd(ad + 2, __floats2half2_rn(v1.x, v1.y));
  atomicAdd(ad + 3, __floats2half2_rn(v1.z, v1.w));
  if (c == 0) atomicAdd(deg + dst, 1.0f);
}

// out = tanh((agg/max(deg,1)) @ Wl + x @ Wr + b); one wave per 16-row tile.
// Weights TDM-staged to LDS once per workgroup.
__global__ __launch_bounds__(256)
void sage_gemm_kernel(const float* __restrict__ x, const _Float16* __restrict__ agg,
                      const float* __restrict__ deg,
                      const float* __restrict__ Wl, const float* __restrict__ Wr,
                      const float* __restrict__ bias,
                      float* __restrict__ out, int n) {
  __shared__ float wsh[2][HID * HID];
  int lane = threadIdx.x & 31;
  int wid  = threadIdx.x >> 5;
  int wave = (int)((blockIdx.x * blockDim.x + threadIdx.x) >> 5);
  int row0 = wave << 4;

  if (wid == 0) {
    tdm_load_2d_f32(Wl, &wsh[0][0], HID, HID);
    tdm_load_2d_f32(Wr, &wsh[1][0], HID, HID);
    __builtin_amdgcn_s_wait_tensorcnt(0);
  }
  __syncthreads();
  if (row0 >= n) return;          // no barriers past this point

  int r = lane & 15;
  float s = 1.0f / fmaxf(deg[row0 + r], 1.0f);
  int koff = (lane >> 4) << 3;
  const float*    xp = x   + (size_t)(row0 + r) * HID;
  const _Float16* ap = agg + (size_t)(row0 + r) * HID;
  v16h aX, aG;
#pragma unroll
  for (int i = 0; i < 8; ++i) {
    aX[i]     = (_Float16)xp[koff + i];
    aX[i + 8] = (_Float16)xp[16 + koff + i];
    aG[i]     = (_Float16)((float)ap[koff + i] * s);
    aG[i + 8] = (_Float16)((float)ap[16 + koff + i] * s);
  }
#pragma unroll
  for (int nt = 0; nt < 2; ++nt) {
    v16h bL = load_b(&wsh[0][0], HID, nt * 16, lane, 32);
    v16h bR = load_b(&wsh[1][0], HID, nt * 16, lane, 32);
    float bv = bias[nt * 16 + (lane & 15)];
    v8f c;
#pragma unroll
    for (int q = 0; q < 8; ++q) c[q] = bv;
    c = __builtin_amdgcn_wmma_f32_16x16x32_f16(false, aG, false, bL, (short)0, c, false, false);
    c = __builtin_amdgcn_wmma_f32_16x16x32_f16(false, aX, false, bR, (short)0, c, false, false);
    store_c_tanh(out, HID, row0, nt * 16, lane, c);
  }
}

// Fused MLP tail 32->16->32->32->2. TDM-staged weights; per-wave LDS slice
// for the C->A transpose (in-order DS per wave + s_wait_dscnt fence).
__global__ __launch_bounds__(256)
void mlp_kernel(const float* __restrict__ x,
                const float* __restrict__ W1, const float* __restrict__ b1,
                const float* __restrict__ W2, const float* __restrict__ b2,
                const float* __restrict__ W3, const float* __restrict__ b3,
                const float* __restrict__ Wf, const float* __restrict__ bf,
                float* __restrict__ y, int n) {
  __shared__ float wsh[2112];          // W1@0(512) W2@512(512) W3@1024(1024) Wf@2048(64)
  __shared__ float hbuf[8][768];       // per-wave: h1[16x16]@0, h2[16x32]@256
  int lane = threadIdx.x & 31;
  int wid  = threadIdx.x >> 5;
  int wave = (int)((blockIdx.x * blockDim.x + threadIdx.x) >> 5);
  int row0 = wave << 4;

  if (wid == 0) {
    tdm_load_2d_f32(W1, &wsh[0],    32, 16);
    tdm_load_2d_f32(W2, &wsh[512],  16, 32);
    tdm_load_2d_f32(W3, &wsh[1024], 32, 32);
    tdm_load_2d_f32(Wf, &wsh[2048], 32, 2);
    __builtin_amdgcn_s_wait_tensorcnt(0);
  }
  __syncthreads();
  if (row0 >= n) return;          // no barriers past this point

  float* h = &hbuf[wid][0];
  int rA = lane & 15;
  int koff = (lane >> 4) << 3;
  int ncol = lane & 15;
  int mhi  = (lane >> 4) << 3;

  // layer 1: [16x32] @ W1[32x16]
  v16h a0 = load_a32(x, row0, HID, lane);
  {
    v16h bW = load_b(&wsh[0], 16, 0, lane, 32);
    float bv = b1[ncol];
    v8f c;
#pragma unroll
    for (int q = 0; q < 8; ++q) c[q] = bv;
    c = __builtin_amdgcn_wmma_f32_16x16x32_f16(false, a0, false, bW, (short)0, c, false, false);
#pragma unroll
    for (int r = 0; r < 8; ++r) h[(mhi + r) * 16 + ncol] = htanh(c[r]);
  }
  asm volatile("s_wait_dscnt 0" ::: "memory");

  // layer 2: [16x16] (K zero-padded to 32) @ W2[16x32]
  v16h a1;
  {
    const float* hp = h + rA * 16;
#pragma unroll
    for (int i = 0; i < 8; ++i) { a1[i] = (_Float16)hp[koff + i]; a1[i + 8] = (_Float16)0.0f; }
  }
#pragma unroll
  for (int nt = 0; nt < 2; ++nt) {
    v16h bW = load_b(&wsh[512], 32, nt * 16, lane, 16);
    float bv = b2[nt * 16 + ncol];
    v8f c;
#pragma unroll
    for (int q = 0; q < 8; ++q) c[q] = bv;
    c = __builtin_amdgcn_wmma_f32_16x16x32_f16(false, a1, false, bW, (short)0, c, false, false);
#pragma unroll
    for (int r = 0; r < 8; ++r) h[256 + (mhi + r) * 32 + nt * 16 + ncol] = htanh(c[r]);
  }
  asm volatile("s_wait_dscnt 0" ::: "memory");

  // layer 3: [16x32] @ W3[32x32]
  v16h a2;
  {
    const float* hp = h + 256 + rA * 32;
#pragma unroll
    for (int i = 0; i < 8; ++i) { a2[i] = (_Float16)hp[koff + i]; a2[i + 8] = (_Float16)hp[16 + koff + i]; }
  }
#pragma unroll
  for (int nt = 0; nt < 2; ++nt) {
    v16h bW = load_b(&wsh[1024], 32, nt * 16, lane, 32);
    float bv = b3[nt * 16 + ncol];
    v8f c;
#pragma unroll
    for (int q = 0; q < 8; ++q) c[q] = bv;
    c = __builtin_amdgcn_wmma_f32_16x16x32_f16(false, a2, false, bW, (short)0, c, false, false);
#pragma unroll
    for (int r = 0; r < 8; ++r) h[(mhi + r) * 32 + nt * 16 + ncol] = htanh(c[r]);  // h3 @0
  }
  asm volatile("s_wait_dscnt 0" ::: "memory");

  // final: h3[16x32] @ Wf[32x2] + bf; lane -> (row = lane&15, col = lane>>4)
  {
    int row = lane & 15, col = lane >> 4;
    float acc = bf[col];
    const float* hp = h + row * 32;
    const float* wf = &wsh[2048];
#pragma unroll
    for (int k = 0; k < 32; ++k) acc += hp[k] * wf[k * 2 + col];
    y[(size_t)(row0 + row) * 2 + col] = acc;
  }
}

// --- QR of [n,2]: Gram matrix reductions, then Gram-Schmidt rescale --------

__global__ __launch_bounds__(256)
void gram_kernel(const float* __restrict__ y, int n, float* __restrict__ scal) {
  float s00 = 0.0f, s01 = 0.0f, s11 = 0.0f;
  for (int i = blockIdx.x * blockDim.x + threadIdx.x; i < n; i += gridDim.x * blockDim.x) {
    float c0 = y[2 * i], c1 = y[2 * i + 1];
    s00 += c0 * c0; s01 += c0 * c1; s11 += c1 * c1;
  }
#pragma unroll
  for (int off = 16; off > 0; off >>= 1) {
    s00 += __shfl_xor(s00, off, 32);
    s01 += __shfl_xor(s01, off, 32);
    s11 += __shfl_xor(s11, off, 32);
  }
  if ((threadIdx.x & 31) == 0) {
    atomicAdd(&scal[0], s00);
    atomicAdd(&scal[1], s01);
    atomicAdd(&scal[2], s11);
  }
}

__global__ void qr_finalize_kernel(float* __restrict__ scal) {
  if (threadIdx.x == 0) {
    float s00 = scal[0], s01 = scal[1], s11 = scal[2];
    float r11 = sqrtf(s00);
    float inv11 = 1.0f / r11;
    float r12 = s01 * inv11;
    float r22 = sqrtf(fmaxf(s11 - r12 * r12, 0.0f));
    scal[3] = inv11;
    scal[4] = r12 * inv11;
    scal[5] = 1.0f / r22;
  }
}

__global__ __launch_bounds__(256)
void qr_apply_kernel(const float* __restrict__ y, const float* __restrict__ scal,
                     float* __restrict__ out, int n) {
  int i = blockIdx.x * blockDim.x + threadIdx.x;
  if (i >= n) return;
  float inv11 = scal[3], t = scal[4], inv22 = scal[5];
  float c0 = y[2 * i], c1 = y[2 * i + 1];
  out[2 * i]     = c0 * inv11;
  out[2 * i + 1] = (c1 - t * c0) * inv22;
}

// --------------------------- host orchestration ----------------------------

extern "C" void kernel_launch(void* const* d_in, const int* in_sizes, int n_in,
                              void* d_out, int out_size, void* d_ws, size_t ws_size,
                              hipStream_t stream) {
  (void)in_sizes; (void)n_in; (void)out_size; (void)ws_size;
  static const int LS[7] = {524288, 65536, 8192, 1024, 128, 16, 2};

  const float* Wc_l = (const float*)d_in[13];
  const float* Wc_r = (const float*)d_in[14];
  const float* bc   = (const float*)d_in[15];
  const float* Wp_l[2] = {(const float*)d_in[16], (const float*)d_in[19]};
  const float* Wp_r[2] = {(const float*)d_in[17], (const float*)d_in[20]};
  const float* bp[2]   = {(const float*)d_in[18], (const float*)d_in[21]};
  const float* W1 = (const float*)d_in[22]; const float* b1 = (const float*)d_in[23];
  const float* W2 = (const float*)d_in[24]; const float* b2 = (const float*)d_in[25];
  const float* W3 = (const float*)d_in[26]; const float* b3 = (const float*)d_in[27];
  const float* Wf = (const float*)d_in[28]; const float* bf = (const float*)d_in[29];

  const size_t BIG = (size_t)524288 * HID * sizeof(float);   // 64 MiB
  char* ws = (char*)d_ws;
  float* bufA = (float*)(ws);
  float* bufB = (float*)(ws + BIG);
  float* bufC = (float*)(ws + 2 * BIG);                      // f16 agg, later f32 y
  float* degb = (float*)(ws + 3 * BIG);
  float* scal = (float*)(ws + 3 * BIG + (size_t)524288 * sizeof(float));

  coarse_kernel<<<1, 32, 0, stream>>>(Wc_l, Wc_r, bc, bufA);

  float* cur = bufA;
  for (int l = 5; l >= 0; --l) {
    int n = LS[l];
    int E = n * 16;
    const int* edges = (const int*)d_in[2 * l];
    const int* inv   = (const int*)d_in[2 * l + 1];
    float* g = (cur == bufA) ? bufB : bufA;

    gather_kernel<<<(unsigned)((n * 8 + 255) / 256), 256, 0, stream>>>(cur, inv, g, n);

    for (int p = 0; p < 2; ++p) {
      const float* in_x = (p == 0) ? g : cur;
      float* out_x      = (p == 0) ? cur : g;
      // zero f16 agg (n*32 halfs = n*16 f32) and f32 deg
      zero_kernel<<<(unsigned)(((long long)n * 16 + 255) / 256), 256, 0, stream>>>(bufC, (long long)n * 16);
      zero_kernel<<<(unsigned)((n + 255) / 256), 256, 0, stream>>>(degb, n);
      aggregate_kernel<<<(unsigned)(((long long)E * 4 + 255) / 256), 256, 0, stream>>>(
          in_x, edges, E, (__half2*)bufC, degb);
      int waves = n / 16;
      sage_gemm_kernel<<<(unsigned)((waves * 32 + 255) / 256), 256, 0, stream>>>(
          in_x, (const _Float16*)bufC, degb, Wp_l[p], Wp_r[p], bp[p], out_x, n);
    }
    cur = g;
  }

  int n0 = LS[0];
  float* y = bufC;   // agg buffer dead now
  {
    int waves = n0 / 16;
    mlp_kernel<<<(unsigned)((waves * 32 + 255) / 256), 256, 0, stream>>>(
        cur, W1, b1, W2, b2, W3, b3, Wf, bf, y, n0);
  }

  zero_kernel<<<1, 32, 0, stream>>>(scal, 8);
  gram_kernel<<<2048, 256, 0, stream>>>(y, n0, scal);
  qr_finalize_kernel<<<1, 1, 0, stream>>>(scal);
  qr_apply_kernel<<<(unsigned)((n0 + 255) / 256), 256, 0, stream>>>(y, scal, (float*)d_out, n0);
}
